// GCN_13615046328635
// MI455X (gfx1250) — matrix operand order repaired and checked by reading, not compiled, not run on hardware
//
#include <hip/hip_runtime.h>

// ---------------------------------------------------------------------------
// GCN on MI455X (gfx1250): bf16 WMMA (V_WMMA_F32_16X16X32_BF16) GEMMs with
// f32 accumulation. Adjacency row-normalization is folded into the propagation
// epilogue (diag(s)*(A@Z) == (diag(s)A)@Z); adjacency is stored once as
// unnormalized bf16 (134 MB -> L2-resident across all 3 steps, 192 MB L2).
//
// B-operand tiles are staged LDS via GLOBAL_LOAD_ASYNC_TO_LDS_B128 (ASYNCcnt),
// double-buffered so the async DMA of tile k+1 overlaps the WMMA chain of
// tile k. One workgroup barrier per K-step.
// ---------------------------------------------------------------------------

typedef __attribute__((ext_vector_type(16))) __bf16 v16bf;
typedef __attribute__((ext_vector_type(8)))  float  v8f;
typedef __bf16 bf16;

#define B_BLK   256
#define NNODE   512
#define DIN     256
#define DHID    256
#define DOUT    128

__device__ __forceinline__ void async_ld_b128(unsigned lds_addr, const void* gp) {
    unsigned long long ga = (unsigned long long)gp;
    asm volatile("global_load_async_to_lds_b128 %0, %1, off"
                 :: "v"(lds_addr), "v"(ga) : "memory");
}
__device__ __forceinline__ void wait_async0() {
    asm volatile("s_wait_asynccnt 0x0" ::: "memory");
}

// ---------------- weight transpose + bf16 convert ---------------------------
__global__ __launch_bounds__(256) void k_weights(const float* __restrict__ W0,
                                                 const float* __restrict__ W1,
                                                 const float* __restrict__ W2,
                                                 bf16* __restrict__ Wt) {
    int idx   = blockIdx.x * 256 + threadIdx.x;          // 3*65536 total
    int which = idx >> 16;
    int rem   = idx & 65535;
    int n     = rem >> 8;        // h_out
    int k     = rem & 255;       // h_in
    const float* W = (which == 0) ? W0 : (which == 1 ? W1 : W2);
    Wt[which * 65536 + n * 256 + k] = (bf16)W[k * 256 + n];
}

// ---------------- feat -> bf16 X, plus nonzero-row flags --------------------
__global__ __launch_bounds__(256) void k_feat(const float* __restrict__ feat,
                                              bf16* __restrict__ X,
                                              float* __restrict__ nz) {
    int wave = threadIdx.x >> 5;
    int lane = threadIdx.x & 31;
    long row = (long)blockIdx.x * 8 + wave;              // 131072 rows
    const float* src = feat + row * DIN + lane * 8;
    float4 a = ((const float4*)src)[0];
    float4 b = ((const float4*)src)[1];
    int flag = (a.x != 0.f) | (a.y != 0.f) | (a.z != 0.f) | (a.w != 0.f) |
               (b.x != 0.f) | (b.y != 0.f) | (b.z != 0.f) | (b.w != 0.f);
    #pragma unroll
    for (int o = 16; o > 0; o >>= 1) flag |= __shfl_xor(flag, o, 32);

    union { bf16 h[8]; uint4 u; } pk;
    pk.h[0] = (bf16)a.x; pk.h[1] = (bf16)a.y; pk.h[2] = (bf16)a.z; pk.h[3] = (bf16)a.w;
    pk.h[4] = (bf16)b.x; pk.h[5] = (bf16)b.y; pk.h[6] = (bf16)b.z; pk.h[7] = (bf16)b.w;
    *(uint4*)(X + row * DIN + lane * 8) = pk.u;
    if (lane == 0) nz[row] = flag ? 1.0f : 0.0f;
}

// ------------- adjacency: masked row sums (f32) + bf16 convert --------------
__global__ __launch_bounds__(256) void k_adj(const float* __restrict__ raw,
                                             const float* __restrict__ nz,
                                             bf16* __restrict__ adjB,
                                             float* __restrict__ inv) {
    int wave = threadIdx.x >> 5;
    int lane = threadIdx.x & 31;
    long rid = (long)blockIdx.x * 8 + wave;              // (b,i) flat, 131072
    long bBlk = rid >> 9;
    const float* rsrc = raw + rid * NNODE + lane * 16;
    const float* nsrc = nz + bBlk * NNODE + lane * 16;

    float sum = 0.f;
    union { bf16 h[8]; uint4 u; } lo, hi;
    #pragma unroll
    for (int c = 0; c < 4; ++c) {
        float4 r = ((const float4*)rsrc)[c];
        float4 m = ((const float4*)nsrc)[c];
        sum += r.x * m.x + r.y * m.y + r.z * m.z + r.w * m.w;
        bf16* dst = (c < 2) ? &lo.h[c * 4] : &hi.h[(c - 2) * 4];
        dst[0] = (bf16)r.x; dst[1] = (bf16)r.y; dst[2] = (bf16)r.z; dst[3] = (bf16)r.w;
    }
    bf16* out = adjB + rid * NNODE + lane * 16;
    ((uint4*)out)[0] = lo.u;
    ((uint4*)out)[1] = hi.u;

    #pragma unroll
    for (int o = 16; o > 0; o >>= 1) sum += __shfl_xor(sum, o, 32);
    if (lane == 0) inv[rid] = (sum == 0.f) ? 1.0f : (1.0f / sum);
}

// ---------------- WMMA bf16 GEMM:  D(MxN) = A(MxK) @ B(KxN) -----------------
// A: row-major, K-contiguous (per-lane b128 pairs).  B: source row n holds
// contiguous K (Bm + n*ldb + k); K=32 x N=128 slabs staged into LDS with
// async-to-LDS DMA, double-buffered.  8 waves/WG, wave w owns rows
// [m0+16w,+16), all waves share the N=128 strip.
// MODE 0: store bf16 to C.   MODE 1: relu(inv[m]*acc) -> bf16.
template <int MODE>
__global__ __launch_bounds__(256)
void gemm_bf16(const bf16* __restrict__ A, long aBatch, int lda,
               const bf16* __restrict__ Bm, long bBatch, int ldb,
               bf16* __restrict__ C, long cBatch, int ldc,
               int K, const float* __restrict__ invScale) {
    __shared__ alignas(16) unsigned char ldsB[2][128 * 80]; // 80B pitch: bank-spread
    const int b    = blockIdx.z;
    const int n0   = blockIdx.x * 128;
    const int m0   = blockIdx.y * 128;
    const int tid  = threadIdx.x;
    const int wave = tid >> 5;
    const int lane = tid & 31;
    const int half = lane >> 4;
    const int l16  = lane & 15;

    const bf16* Ab = A + (long)b * aBatch;
    const bf16* Bb = Bm + (long)b * bBatch;
    bf16*       Cb = C + (long)b * cBatch;

    const int   mWave = m0 + wave * 16;
    const bf16* aRow  = Ab + (long)(mWave + l16) * lda;

    // Per-thread staging descriptors: 512 b128 chunks/tile, 2 per thread.
    const int r0 = tid >> 2,          p0 = tid & 3;
    const int r1 = (tid + 256) >> 2,  p1 = (tid + 256) & 3;
    const bf16* g0 = Bb + (long)(n0 + r0) * ldb + p0 * 8;   // + kt*32 later
    const bf16* g1 = Bb + (long)(n0 + r1) * ldb + p1 * 8;
    const unsigned BUFSZ = 128 * 80;
    const unsigned l0 = (unsigned)(unsigned long long)&ldsB[0][r0 * 80 + p0 * 16];
    const unsigned l1 = (unsigned)(unsigned long long)&ldsB[0][r1 * 80 + p1 * 16];

    v8f acc[8] = {};
    const int nsteps = K >> 5;

    // Prologue: DMA tile 0 into buffer 0.
    async_ld_b128(l0, g0);
    async_ld_b128(l1, g1);
    wait_async0();
    __syncthreads();

    for (int kt = 0; kt < nsteps; ++kt) {
        const int cur = kt & 1;
        // Kick off async DMA of the next K-slab into the other buffer.
        if (kt + 1 < nsteps) {
            const int koff = (kt + 1) * 32;
            const unsigned bsel = (cur ^ 1) * BUFSZ;
            async_ld_b128(l0 + bsel, g0 + koff);
            async_ld_b128(l1 + bsel, g1 + koff);
        }

        // A fragment: lane = row m, chunks [k+8*half,+8) and +16.
        union { uint4 u[2]; v16bf v; } aF;
        const uint4* ap = (const uint4*)(aRow + kt * 32 + 8 * half);
        aF.u[0] = ap[0];
        aF.u[1] = ap[2];                       // +16 elems = +2 uint4
        if (kt + 1 < nsteps) __builtin_prefetch(aRow + kt * 32 + 32, 0, 3);

        // Preload ALL 8 B fragments (batched ds_load_b128), then WMMA chain.
        union { uint4 u[2]; v16bf v; } bF[8];
        #pragma unroll
        for (int t = 0; t < 8; ++t) {
            const uint4* bp =
                (const uint4*)(&ldsB[cur][(t * 16 + l16) * 80 + half * 16]);
            bF[t].u[0] = bp[0];
            bF[t].u[1] = bp[2];
        }
        #pragma unroll
        for (int t = 0; t < 8; ++t) {
            acc[t] = __builtin_amdgcn_wmma_f32_16x16x32_bf16(
                false, aF.v, false, bF[t].v, (short)0, acc[t], false, false);
        }

        wait_async0();       // own DMA for next buffer done
        __syncthreads();     // everyone's DMA done; everyone done reading cur
    }

    // Epilogue: C tile lane = fixed n, rows m = mWave + 8*half + r.
    float sc[8];
    if (MODE == 1) {
        #pragma unroll
        for (int r = 0; r < 8; ++r)
            sc[r] = invScale[(long)b * NNODE + mWave + 8 * half + r];
    }
    #pragma unroll
    for (int t = 0; t < 8; ++t) {
        int n = n0 + t * 16 + l16;
        #pragma unroll
        for (int r = 0; r < 8; ++r) {
            int   m = mWave + 8 * half + r;
            float v = acc[t][r];
            if (MODE == 1) v = fmaxf(v * sc[r], 0.0f);
            Cb[(long)m * ldc + n] = (bf16)v;
        }
    }
}

// ---------------- output projection of node 0 (f32 weights) -----------------
__global__ __launch_bounds__(128) void k_out(const bf16* __restrict__ X,
                                             const float* __restrict__ Wout,
                                             float* __restrict__ out) {
    int b = blockIdx.x, o = threadIdx.x;
    const bf16* x = X + (long)b * (NNODE * DHID);    // node 0 row
    float s = 0.f;
    #pragma unroll 8
    for (int h = 0; h < DHID; ++h) s += (float)x[h] * Wout[h * DOUT + o];
    out[b * DOUT + o] = s;
}

// ---------------------------------------------------------------------------
extern "C" void kernel_launch(void* const* d_in, const int* in_sizes, int n_in,
                              void* d_out, int out_size, void* d_ws, size_t ws_size,
                              hipStream_t stream) {
    const float* feat = (const float*)d_in[0];
    const float* raw  = (const float*)d_in[1];
    const float* W0   = (const float*)d_in[2];
    const float* W1   = (const float*)d_in[3];
    const float* W2   = (const float*)d_in[4];
    const float* Wout = (const float*)d_in[5];
    float* out = (float*)d_out;

    // Workspace layout (~257.5 MiB total)
    char* ws = (char*)d_ws;
    bf16*  adjB = (bf16*)(ws);                         // 134,217,728 B
    bf16*  X    = (bf16*)(ws + 134217728ull);          //  67,108,864 B
    bf16*  Zt   = (bf16*)(ws + 201326592ull);          //  67,108,864 B
    bf16*  Wt   = (bf16*)(ws + 268435456ull);          //     393,216 B
    float* nz   = (float*)(ws + 268828672ull);         //     524,288 B
    float* inv  = (float*)(ws + 269352960ull);         //     524,288 B

    k_weights<<<768, 256, 0, stream>>>(W0, W1, W2, Wt);
    k_feat<<<16384, 256, 0, stream>>>(feat, X, nz);
    k_adj<<<16384, 256, 0, stream>>>(raw, nz, adjB, inv);

    for (int s = 0; s < 3; ++s) {
        // Z^T = W^T @ X^T : M=256 (h_out), N=512 (node), K=256 (h_in)
        gemm_bf16<0><<<dim3(4, 2, B_BLK), 256, 0, stream>>>(
            Wt + s * 65536, 0, 256,
            X, (long)NNODE * DHID, 256,
            Zt, (long)DHID * NNODE, 512,
            256, nullptr);
        // X' = relu(inv .* (A @ Z)) : M=512 (node), N=256 (h), K=512 (node)
        gemm_bf16<1><<<dim3(2, 4, B_BLK), 256, 0, stream>>>(
            adjB, (long)NNODE * NNODE, 512,
            Zt, (long)DHID * NNODE, 512,
            X, (long)NNODE * DHID, 256,
            512, inv);
    }

    k_out<<<B_BLK, 128, 0, stream>>>(X, Wout, out);
}
